// PSN_34342558499170
// MI455X (gfx1250) — matrix operand order repaired and checked by reading, not compiled, not run on hardware
//
#include <hip/hip_runtime.h>
#include <hip/hip_bf16.h>
#include <math.h>

// ---------------------------------------------------------------------------
// Problem constants (from reference): N=10000 nodes, E=320000 edges, F=128,
// L=8 layers.  Propagation phase is L2-atomic bound; final GEMM uses WMMA f16.
// ---------------------------------------------------------------------------

typedef __attribute__((ext_vector_type(16))) _Float16 v16h;
typedef __attribute__((ext_vector_type(8)))  _Float16 v8h;
typedef __attribute__((ext_vector_type(8)))  float    v8f;

#define F_DIM 128

// ---------------------------------------------------------------------------
// Elementwise / setup kernels
// ---------------------------------------------------------------------------

__global__ void k_zero_f32(float* __restrict__ p, int n) {
  int i = blockIdx.x * blockDim.x + threadIdx.x;
  if (i < n) p[i] = 0.0f;
}

__global__ void k_copy_f32(const float* __restrict__ s, float* __restrict__ d, int n4) {
  int i = blockIdx.x * blockDim.x + threadIdx.x;
  if (i < n4) {
    float4 v = ((const float4*)s)[i];
    ((float4*)d)[i] = v;
  }
}

__global__ void k_degree(const int* __restrict__ col, float* __restrict__ deg, int E) {
  int e = blockIdx.x * blockDim.x + threadIdx.x;
  if (e < E) {
    unsafeAtomicAdd(&deg[col[e]], 1.0f);
  }
}

__global__ void k_deg_inv_sqrt(float* __restrict__ deg, int n) {
  int i = blockIdx.x * blockDim.x + threadIdx.x;
  if (i < n) {
    float d = deg[i];
    d = (d == 0.0f) ? 1.0f : d;
    deg[i] = rsqrtf(d);
  }
}

__global__ void k_edge_norm(const int* __restrict__ row, const int* __restrict__ col,
                            const float* __restrict__ dinv, float* __restrict__ norm, int E) {
  int e = blockIdx.x * blockDim.x + threadIdx.x;
  if (e < E) {
    norm[e] = dinv[row[e]] * dinv[col[e]];
  }
}

// ---------------------------------------------------------------------------
// Scatter: one edge per wave32; each lane owns 4 contiguous features.
// float4 gather from cur[row], 4x global_atomic_add_f32 into agg[col].
// Working set (~25 MB) is resident in the 192 MB L2, so these atomics are
// L2-local RMWs, the fastest available path for unsorted edge lists.
// ---------------------------------------------------------------------------
__global__ void k_scatter(const int* __restrict__ row, const int* __restrict__ col,
                          const float* __restrict__ norm, const float* __restrict__ cur,
                          float* __restrict__ agg, int E) {
  int e = blockIdx.x * (blockDim.x >> 5) + (threadIdx.x >> 5);
  int lane = threadIdx.x & 31;
  if (e >= E) return;
  int r = row[e];
  int c = col[e];
  float nm = norm[e];
  const float4* __restrict__ src = (const float4*)(cur + (size_t)r * F_DIM);
  float4 v = src[lane];
  float* __restrict__ dst = agg + (size_t)c * F_DIM + lane * 4;
  unsafeAtomicAdd(dst + 0, nm * v.x);
  unsafeAtomicAdd(dst + 1, nm * v.y);
  unsafeAtomicAdd(dst + 2, nm * v.z);
  unsafeAtomicAdd(dst + 3, nm * v.w);
}

// cur -= agg; update += tanh(k[layer]) * cur   (vectorized float4)
__global__ void k_layer_update(float* __restrict__ cur, const float* __restrict__ agg,
                               float* __restrict__ upd, const float* __restrict__ kvals,
                               int layer, int n4) {
  int i = blockIdx.x * blockDim.x + threadIdx.x;
  if (i >= n4) return;
  float tk = tanhf(kvals[layer]);
  float4 c = ((const float4*)cur)[i];
  float4 a = ((const float4*)agg)[i];
  c.x -= a.x; c.y -= a.y; c.z -= a.z; c.w -= a.w;
  ((float4*)cur)[i] = c;
  float4 u = ((const float4*)upd)[i];
  u.x += tk * c.x; u.y += tk * c.y; u.z += tk * c.z; u.w += tk * c.w;
  ((float4*)upd)[i] = u;
}

// h = sigmoid(w)*update + (1-sigmoid(w))*x, stored as f16 for WMMA
__global__ void k_make_h16(const float* __restrict__ x, const float* __restrict__ upd,
                           const float* __restrict__ wgt, _Float16* __restrict__ h16, int n) {
  int i = blockIdx.x * blockDim.x + threadIdx.x;
  if (i >= n) return;
  float cst = 1.0f / (1.0f + __expf(-wgt[0]));
  h16[i] = (_Float16)(cst * upd[i] + (1.0f - cst) * x[i]);
}

__global__ void k_cvt_w16(const float* __restrict__ W, _Float16* __restrict__ w16, int n) {
  int i = blockIdx.x * blockDim.x + threadIdx.x;
  if (i < n) w16[i] = (_Float16)W[i];
}

// ---------------------------------------------------------------------------
// Final GEMM: out = relu(h @ W.T + b), h:[N,128] f16, W:[128,128] f16.
// One wave32 per 16x16 output tile.  K-loop: 4 x v_wmma_f32_16x16x32_f16.
//
// Fragment layouts per CDNA5 ISA 7.12.2 (wave32):
//   A (16x32 f16): lane L, half-index i -> M = L&15, K = i + 8*(i>=8) + 8*(L>=16)
//     => two contiguous 8-half (16B) loads at K offsets {sel*8, sel*8+16}
//   B (32x16 f16): lane L, half-index i -> N = L&15, K = i + 8*(i>=8) + 8*(L>=16)
//     => two contiguous 8-half (16B) loads at K offsets {sel*8, sel*8+16}
//   C/D (16x16 f32, 8 VGPRs): lane L, vgpr v -> M = v + 8*(L>=16), N = L&15
//
// Epilogue: single uniform full-tile test; common path is branch-free with
// 8 immediate-offset stores (row stride 512 B < 24-bit IOFFSET).
// ---------------------------------------------------------------------------
__global__ void k_gemm_wmma(const _Float16* __restrict__ h16,
                            const _Float16* __restrict__ w16,
                            const float* __restrict__ bias,
                            float* __restrict__ out, int Nrows) {
  const int mTile = blockIdx.x;      // node-row tile (16 rows)
  const int nTile = blockIdx.y;      // output-column tile (16 cols), 0..7
  const int lane  = threadIdx.x;     // 0..31
  const int sel   = lane >> 4;       // 0 / 1
  const int l15   = lane & 15;

  int aRow = mTile * 16 + l15;           // A-matrix row (M)
  if (aRow >= Nrows) aRow = Nrows - 1;   // clamp loads for ragged tail
  const int bRow = nTile * 16 + l15;     // W row == output column j (B is W^T)

  const _Float16* __restrict__ aBase = h16 + (size_t)aRow * F_DIM;
  const _Float16* __restrict__ bBase = w16 + (size_t)bRow * F_DIM;

  v8f acc = {};
#pragma unroll
  for (int kc = 0; kc < F_DIM; kc += 32) {
    v8h a0 = *(const v8h*)(aBase + kc + sel * 8);
    v8h a1 = *(const v8h*)(aBase + kc + sel * 8 + 16);
    v16h A = __builtin_shufflevector(a0, a1, 0,1,2,3,4,5,6,7,8,9,10,11,12,13,14,15);

    v8h b0 = *(const v8h*)(bBase + kc + sel * 8);
    v8h b1 = *(const v8h*)(bBase + kc + sel * 8 + 16);
    v16h B = __builtin_shufflevector(b0, b1, 0,1,2,3,4,5,6,7,8,9,10,11,12,13,14,15);

    acc = __builtin_amdgcn_wmma_f32_16x16x32_f16(
        /*neg_a=*/false, A, /*neg_b=*/false, B,
        /*c_mod=*/(short)0, acc, /*reuse_a=*/false, /*reuse_b=*/false);
  }

  const float bv = bias[nTile * 16 + l15];
  const int   mBase = mTile * 16 + sel * 8;               // first row this lane stores
  float* __restrict__ oBase = out + (size_t)mBase * F_DIM + nTile * 16 + l15;

  if (mTile * 16 + 16 <= Nrows) {
    // full tile: branch-free, immediate-offset stores
#pragma unroll
    for (int v = 0; v < 8; ++v) {
      float r = acc[v] + bv;
      oBase[(size_t)v * F_DIM] = (r > 0.0f) ? r : 0.0f;
    }
  } else {
    // ragged tail tile
#pragma unroll
    for (int v = 0; v < 8; ++v) {
      if (mBase + v < Nrows) {
        float r = acc[v] + bv;
        oBase[(size_t)v * F_DIM] = (r > 0.0f) ? r : 0.0f;
      }
    }
  }
}

// ---------------------------------------------------------------------------
// Host launcher
// ---------------------------------------------------------------------------
static inline size_t align_up(size_t v, size_t a) { return (v + a - 1) / a * a; }

extern "C" void kernel_launch(void* const* d_in, const int* in_sizes, int n_in,
                              void* d_out, int out_size, void* d_ws, size_t ws_size,
                              hipStream_t stream) {
  const float* x      = (const float*)d_in[0];
  const int*   eidx   = (const int*)d_in[1];     // [2, E] flat
  const float* kvals  = (const float*)d_in[2];
  const float* wgt    = (const float*)d_in[3];
  const float* W      = (const float*)d_in[4];
  const float* bias   = (const float*)d_in[5];

  const int N  = in_sizes[0] / F_DIM;        // 10000
  const int E  = in_sizes[1] / 2;            // 320000
  const int L  = in_sizes[2];                // 8
  const int NF = N * F_DIM;

  const int* row = eidx;
  const int* col = eidx + E;

  // bump-allocate scratch from d_ws
  size_t off = 0;
  char* ws = (char*)d_ws;
  float*    deg  = (float*)(ws + off);    off = align_up(off + (size_t)N  * 4, 256);
  float*    norm = (float*)(ws + off);    off = align_up(off + (size_t)E  * 4, 256);
  float*    cur  = (float*)(ws + off);    off = align_up(off + (size_t)NF * 4, 256);
  float*    agg  = (float*)(ws + off);    off = align_up(off + (size_t)NF * 4, 256);
  float*    upd  = (float*)(ws + off);    off = align_up(off + (size_t)NF * 4, 256);
  _Float16* h16  = (_Float16*)(ws + off); off = align_up(off + (size_t)NF * 2, 256);
  _Float16* w16  = (_Float16*)(ws + off); off = align_up(off + (size_t)F_DIM * F_DIM * 2, 256);
  (void)ws_size; (void)n_in; (void)out_size;

  const int TB = 256;
  const int nf4 = NF / 4;

  // init: deg=0, update=0, cur=x
  k_zero_f32<<<(N  + TB - 1) / TB, TB, 0, stream>>>(deg, N);
  k_zero_f32<<<(NF + TB - 1) / TB, TB, 0, stream>>>(upd, NF);
  k_copy_f32<<<(nf4 + TB - 1) / TB, TB, 0, stream>>>(x, cur, nf4);

  // degree -> deg^-1/2 -> per-edge norm
  k_degree<<<(E + TB - 1) / TB, TB, 0, stream>>>(col, deg, E);
  k_deg_inv_sqrt<<<(N + TB - 1) / TB, TB, 0, stream>>>(deg, N);
  k_edge_norm<<<(E + TB - 1) / TB, TB, 0, stream>>>(row, col, deg, norm, E);

  // propagation layers
  const int wavesPerBlock = TB / 32;                 // 8 edges per block
  const int scatterBlocks = (E + wavesPerBlock - 1) / wavesPerBlock;
  for (int layer = 0; layer < L; ++layer) {
    k_zero_f32<<<(NF + TB - 1) / TB, TB, 0, stream>>>(agg, NF);
    k_scatter<<<scatterBlocks, TB, 0, stream>>>(row, col, norm, cur, agg, E);
    k_layer_update<<<(nf4 + TB - 1) / TB, TB, 0, stream>>>(cur, agg, upd, kvals, layer, nf4);
  }

  // finalize h (f16), convert W (f16), WMMA GEMM + bias + relu
  k_make_h16<<<(NF + TB - 1) / TB, TB, 0, stream>>>(x, upd, wgt, h16, NF);
  k_cvt_w16<<<(F_DIM * F_DIM + TB - 1) / TB, TB, 0, stream>>>(W, w16, F_DIM * F_DIM);

  dim3 gemmGrid((N + 15) / 16, F_DIM / 16);
  k_gemm_wmma<<<gemmGrid, 32, 0, stream>>>(h16, w16, bias, (float*)d_out, N);
}
